// BahdanauAttention_90958817395435
// MI455X (gfx1250) — compile-verified, matrix operand order checked by reading, not run on hardware
//
#include <hip/hip_runtime.h>
#include <hip/hip_bf16.h>
#include <math.h>

// Problem sizes (fixed by the reference): B=64, T=1024, KS=QS=H=1024.
constexpr int BB = 64;
constexpr int TT = 1024;
constexpr int KK = 1024;   // KS (also QS)
constexpr int HH = 1024;

constexpr int MT   = 32;   // rows of T per block (two 16-row WMMA M-tiles)
constexpr int SLAB = 256;  // K-chunk of keys staged in LDS per iteration

typedef __attribute__((ext_vector_type(16))) __bf16 v16bf;
typedef __attribute__((ext_vector_type(8)))  __bf16 v8bf;
typedef __attribute__((ext_vector_type(8)))  float  v8f;

// gfx1250 has V_TANH_F32 (TRANS op). Use the builtin when clang exposes it.
#if defined(__has_builtin)
# if __has_builtin(__builtin_amdgcn_tanhf)
#  define FAST_TANH(x) __builtin_amdgcn_tanhf(x)
# elif __has_builtin(__builtin_amdgcn_tanh_f32)
#  define FAST_TANH(x) __builtin_amdgcn_tanh_f32(x)
# else
#  define FAST_TANH(x) tanhf(x)
# endif
#else
# define FAST_TANH(x) tanhf(x)
#endif

// ---------------------------------------------------------------------------
// Kernel 0: transpose + convert W_key [K][H] (f32) -> WT [H][K] (bf16).
// Done once; makes WMMA B-fragment loads contiguous 16-byte chunks.
// ---------------------------------------------------------------------------
__global__ void __launch_bounds__(256)
wkey_transpose_bf16(const float* __restrict__ W, unsigned short* __restrict__ WTraw) {
    __shared__ float tile[32][33];                 // +1 pad: no LDS bank conflicts
    __bf16* WT = reinterpret_cast<__bf16*>(WTraw);
    const int tx = threadIdx.x & 31;
    const int ty = threadIdx.x >> 5;               // 0..7
    const int k0 = blockIdx.x * 32;
    const int h0 = blockIdx.y * 32;
#pragma unroll
    for (int i = 0; i < 32; i += 8)
        tile[ty + i][tx] = W[(size_t)(k0 + ty + i) * HH + (h0 + tx)];
    __syncthreads();
#pragma unroll
    for (int i = 0; i < 32; i += 8)
        WT[(size_t)(h0 + ty + i) * KK + (k0 + tx)] = (__bf16)tile[tx][ty + i];
}

// ---------------------------------------------------------------------------
// Kernel 1: q[b][h] = sum_k query[b][k] * W_query[k][h].  (tiny: 134 MFLOP)
// ---------------------------------------------------------------------------
__global__ void __launch_bounds__(256)
query_proj(const float* __restrict__ query, const float* __restrict__ Wq,
           float* __restrict__ qbuf) {
    const int b = blockIdx.x;
    const int h = blockIdx.y * 256 + threadIdx.x;
    const float* qrow = query + (size_t)b * KK;
    float acc = 0.f;
#pragma unroll 4
    for (int k = 0; k < KK; ++k)
        acc = fmaf(qrow[k], Wq[(size_t)k * HH + h], acc);
    qbuf[(size_t)b * HH + h] = acc;
}

// ---------------------------------------------------------------------------
// Kernel 2 (the heavy one): fused  scores[b][t] =
//     sum_h tanh( q[b][h] + (keys[b] @ W_key)[t][h] ) * w_score[h]
//
// Block = (b, 32-row tile of T). Per K-slab of 256:
//   1) 256 threads cooperatively stage keys[32 x 256] into LDS as bf16,
//      laid out in WMMA A-fragment order (coalesced b128 global reads,
//      one fp32->bf16 convert, contiguous ds stores).
//   2) 8 K-steps of WMMA: each wave owns 8 N-tiles; each B fragment load
//      (global, L2-resident WT) feeds TWO wmmas (both M-tiles).
// proj_keys never touches HBM; accumulators live in VGPRs for the whole loop.
// ---------------------------------------------------------------------------
__global__ void __launch_bounds__(256)
fused_scores(const float* __restrict__ keys,
             const unsigned short* __restrict__ WTraw,
             const float* __restrict__ qbuf,
             const float* __restrict__ w_score,
             float* __restrict__ scores) {
    const __bf16* WT = reinterpret_cast<const __bf16*>(WTraw);
    const int b    = blockIdx.y;
    const int t0   = blockIdx.x * MT;
    const int tid  = threadIdx.x;
    const int wv   = tid >> 5;          // wave id 0..7
    const int lane = tid & 31;
    const int half = lane >> 4;         // 0 or 1 (wave32 half)
    const int rowl = lane & 15;
    const int khalf = half * 8;         // ISA A/B layout: K base per half

    // A staging: [mt][s][lane][e] -> 2*8*32*16 bf16 = 16 KB
    __shared__ __bf16 ldsA[2 * 8 * 32 * 16];
    __shared__ float lds_scores[MT];
    if (tid < MT) lds_scores[tid] = 0.f;

    const float* keybase = keys + ((size_t)b * TT + t0) * KK;

    const v8f vzero = {0.f, 0.f, 0.f, 0.f, 0.f, 0.f, 0.f, 0.f};
    v8f acc[2][8];
#pragma unroll
    for (int mt = 0; mt < 2; ++mt)
#pragma unroll
        for (int j = 0; j < 8; ++j) acc[mt][j] = vzero;

    for (int slab0 = 0; slab0 < KK; slab0 += SLAB) {
        __syncthreads();
        // ---- stage keys slab into LDS in fragment order -------------------
        // 32 rows x 32 chunks-of-8-floats = 1024 chunks; 4 per thread;
        // consecutive tids -> consecutive 32B of the same row (coalesced).
#pragma unroll
        for (int i = 0; i < 4; ++i) {
            const int id   = tid + i * 256;
            const int R    = id >> 5;        // 0..31 (row in tile)
            const int c    = id & 31;        // chunk within row
            const int krel = c * 8;
            const float* g = keybase + (size_t)R * KK + slab0 + krel;
            const float4 g0 = *(const float4*)(g);
            const float4 g1 = *(const float4*)(g + 4);
            if (slab0 + SLAB < KK)
                __builtin_prefetch(g + SLAB, 0, 0);   // global_prefetch_b8
            v8bf v;
            v[0] = (__bf16)g0.x; v[1] = (__bf16)g0.y;
            v[2] = (__bf16)g0.z; v[3] = (__bf16)g0.w;
            v[4] = (__bf16)g1.x; v[5] = (__bf16)g1.y;
            v[6] = (__bf16)g1.z; v[7] = (__bf16)g1.w;
            // fragment position: s = k-step, half/e per 16-bit A layout
            const int mt = R >> 4, r = R & 15;
            const int s  = krel >> 5;
            const int kk = krel & 31;        // 0, 8, 16, 24
            const int h2 = (kk >> 3) & 1;    // half
            const int e0 = (kk & 16) >> 1;   // 0 or 8
            *(v8bf*)(&ldsA[(((mt * 8 + s) * 32 + (h2 * 16 + r)) * 16) + e0]) = v;
        }
        __syncthreads();

        // ---- 8 K-steps of WMMA off the staged slab ------------------------
#pragma unroll 2
        for (int s = 0; s < 8; ++s) {
            const v16bf a0 = *(const v16bf*)(&ldsA[((0 * 8 + s) * 32 + lane) * 16]);
            const v16bf a1 = *(const v16bf*)(&ldsA[((1 * 8 + s) * 32 + lane) * 16]);
            const int kb = slab0 + s * 32 + khalf;
#pragma unroll
            for (int j = 0; j < 8; ++j) {
                const int n = (wv + j * 8) * 16 + rowl;
                const __bf16* wt = WT + (size_t)n * KK + kb;
                const v8bf blo = *(const v8bf*)(wt);
                const v8bf bhi = *(const v8bf*)(wt + 16);
                const v16bf bb = __builtin_shufflevector(
                    blo, bhi, 0, 1, 2, 3, 4, 5, 6, 7,
                    8, 9, 10, 11, 12, 13, 14, 15);
                acc[0][j] = __builtin_amdgcn_wmma_f32_16x16x32_bf16(
                    false, a0, false, bb, (short)0, acc[0][j], false, false);
                acc[1][j] = __builtin_amdgcn_wmma_f32_16x16x32_bf16(
                    false, a1, false, bb, (short)0, acc[1][j], false, false);
            }
        }
    }

    // ---- epilogue: tanh + dot(w_score) + N-reduction ----------------------
    // C layout: lane L, VGPR r holds (M = mt*16 + r + 8*half, N = n0 + rowl).
    float part[2][8];
#pragma unroll
    for (int mt = 0; mt < 2; ++mt)
#pragma unroll
        for (int r = 0; r < 8; ++r) part[mt][r] = 0.f;

#pragma unroll
    for (int j = 0; j < 8; ++j) {
        const int n   = (wv + j * 8) * 16 + rowl;
        const float qv = qbuf[(size_t)b * HH + n];
        const float ws = w_score[n];
#pragma unroll
        for (int mt = 0; mt < 2; ++mt)
#pragma unroll
            for (int r = 0; r < 8; ++r)
                part[mt][r] += FAST_TANH(acc[mt][j][r] + qv) * ws;
    }
#pragma unroll
    for (int mt = 0; mt < 2; ++mt)
#pragma unroll
        for (int r = 0; r < 8; ++r)
            atomicAdd(&lds_scores[mt * 16 + half * 8 + r], part[mt][r]); // ds_add_f32
    __syncthreads();
    if (tid < MT)
        scores[(size_t)b * TT + t0 + tid] = lds_scores[tid];
}

// ---------------------------------------------------------------------------
// Kernel 3: per-batch softmax over T, then context = alphas @ values[b].
// Bandwidth bound: streams values[b] (4 MB) once, coalesced along H.
// ---------------------------------------------------------------------------
__global__ void __launch_bounds__(256)
softmax_context(const float* __restrict__ scores,
                const unsigned char* __restrict__ mask,
                const float* __restrict__ values,
                float* __restrict__ out_ctx,
                float* __restrict__ out_alpha) {
    const int b   = blockIdx.x;
    const int tid = threadIdx.x;
    __shared__ float salpha[TT];
    __shared__ float rbuf[256];

    float s[4];
    float lmax = -INFINITY;
#pragma unroll
    for (int i = 0; i < 4; ++i) {
        const int t = tid + i * 256;
        const float v = mask[(size_t)b * TT + t] ? scores[(size_t)b * TT + t]
                                                 : -INFINITY;
        s[i] = v;
        lmax = fmaxf(lmax, v);
    }
    rbuf[tid] = lmax; __syncthreads();
    for (int off = 128; off > 0; off >>= 1) {
        if (tid < off) rbuf[tid] = fmaxf(rbuf[tid], rbuf[tid + off]);
        __syncthreads();
    }
    const float gmax = rbuf[0];
    __syncthreads();

    float lsum = 0.f;
#pragma unroll
    for (int i = 0; i < 4; ++i) {
        const float e = __expf(s[i] - gmax);
        s[i] = e;
        lsum += e;
    }
    rbuf[tid] = lsum; __syncthreads();
    for (int off = 128; off > 0; off >>= 1) {
        if (tid < off) rbuf[tid] += rbuf[tid + off];
        __syncthreads();
    }
    const float inv = 1.f / rbuf[0];
    __syncthreads();

#pragma unroll
    for (int i = 0; i < 4; ++i) {
        const int t = tid + i * 256;
        const float a = s[i] * inv;
        salpha[t] = a;
        out_alpha[(size_t)b * TT + t] = a;
    }
    __syncthreads();

    for (int d = tid; d < HH; d += 256) {
        float acc = 0.f;
#pragma unroll 4
        for (int t = 0; t < TT; ++t)
            acc = fmaf(salpha[t], values[((size_t)b * TT + t) * HH + d], acc);
        out_ctx[(size_t)b * HH + d] = acc;
    }
}

// ---------------------------------------------------------------------------
// Launcher. Inputs (setup_inputs order): query, mask, values, keys, W_key,
// W_query, w_score. Output: context [64,1,1024] then alphas [64,1,1024].
// Workspace: WT bf16 (2 MB) | qbuf (256 KB) | scores (256 KB).
// ---------------------------------------------------------------------------
extern "C" void kernel_launch(void* const* d_in, const int* in_sizes, int n_in,
                              void* d_out, int out_size, void* d_ws, size_t ws_size,
                              hipStream_t stream) {
    const float*         query   = (const float*)d_in[0];
    const unsigned char* mask    = (const unsigned char*)d_in[1];
    const float*         values  = (const float*)d_in[2];
    const float*         keys    = (const float*)d_in[3];
    const float*         W_key   = (const float*)d_in[4];
    const float*         W_query = (const float*)d_in[5];
    const float*         w_score = (const float*)d_in[6];

    char* ws = (char*)d_ws;
    unsigned short* WT = (unsigned short*)ws;                         // 2 MB bf16
    size_t off = (size_t)HH * KK * sizeof(unsigned short);
    float* qbuf = (float*)(ws + off);                                 // 256 KB
    off += (size_t)BB * HH * sizeof(float);
    float* sbuf = (float*)(ws + off);                                 // 256 KB

    float* out_ctx   = (float*)d_out;
    float* out_alpha = out_ctx + (size_t)BB * HH;

    hipLaunchKernelGGL(wkey_transpose_bf16, dim3(KK / 32, HH / 32), dim3(256),
                       0, stream, W_key, WT);
    hipLaunchKernelGGL(query_proj, dim3(BB, HH / 256), dim3(256),
                       0, stream, query, W_query, qbuf);
    hipLaunchKernelGGL(fused_scores, dim3(TT / MT, BB), dim3(256),
                       0, stream, keys, WT, qbuf, w_score, sbuf);
    hipLaunchKernelGGL(softmax_context, dim3(BB), dim3(256),
                       0, stream, sbuf, mask, values, out_ctx, out_alpha);
}
